// CLIP_DINOiser_68289980007108
// MI455X (gfx1250) — matrix operand order repaired and checked by reading, not compiled, not run on hardware
//
#include <hip/hip_runtime.h>
#include <hip/hip_bf16.h>

// ---------------------------------------------------------------------------
// CLIP_DINOiser pipeline on gfx1250 (CDNA5, wave32, WMMA)
//
//  Stage 0: w_proj f32 -> f16  [256][6912]   (B-matrix for conv GEMM)
//  Stage 1: conv3x3 as im2col WMMA GEMM (M=pixels, N=256 oc, K=6912)
//           + channel L2-norm -> Q f16 [B][PN][256]  (pad rows zeroed)
//  Stage 2: clipmap f32 -> f16 -> V f16 [B][512][PN] (pad cols zeroed)
//  Stage 3: preds (bkg decoder on channel-normed feats) -> d_out tail
//  Per batch:
//    corrs   : S = Q*Qt (WMMA f16->f32, LDS-staged A), threshold 0.2 -> P f16
//    rowsum  : rs[n] = sum_m P[n][m]
//    refined : O = P*Vt (WMMA, unroll x2 + global prefetch),
//              epilogue O/(rs+1e-6) -> d_out [b][512][2401]
// ---------------------------------------------------------------------------

#define HW   2401          // 49*49
#define HH   49
#define PN   2432          // padded hw: 19*128, multiple of 16/32/128
#define KD   256           // proj channels
#define CC   512           // clipmap channels
#define CIN  768
#define KCONV (CIN * 9)    // 6912, im2col K, multiple of 32
#define NB   8
#define LDA  40            // conv LDS A-tile row stride (halves), conflict-free
#define LDQ  264           // corrs LDS A-tile row stride (halves), conflict-free

typedef __attribute__((ext_vector_type(16))) _Float16 v16h;
typedef __attribute__((ext_vector_type(8)))  _Float16 v8h;
typedef __attribute__((ext_vector_type(8)))  float    v8f;

union V16U { v16h v; v8h h[2]; };

// A-fragment (16x32 f16) from row-major source (ld in halves).
// Lane l (l<16): row n0+l, elems 0..15 = K[k..k+15]
// Lane l (>=16): row n0+l-16, elems 0..15 = K[k+16..k+31]
__device__ __forceinline__ v16h load_a_frag(const _Float16* __restrict__ M,
                                            int ld, int n0, int k, int lane) {
  int row = n0 + (lane & 15);
  int kb  = k + ((lane >> 4) << 4);          // +0 or +16
  const _Float16* p = M + (long long)row * ld + kb;
  V16U u;
  u.h[0] = *(const v8h*)(p);
  u.h[1] = *(const v8h*)(p + 8);
  return u.v;
}

// Same A-fragment pattern, from an LDS tile with row stride ld halves.
__device__ __forceinline__ v16h load_a_lds(const _Float16* As, int ld, int k,
                                           int lane) {
  int row = lane & 15;
  int kb  = k + ((lane >> 4) << 4);
  const _Float16* p = As + row * ld + kb;
  V16U u;
  u.h[0] = *(const v8h*)(p);
  u.h[1] = *(const v8h*)(p + 8);
  return u.v;
}

// B-fragment (32x16 f16) from an [N][K] row-major source (K contiguous):
// lane l: column N = n0 + (l&15), 16 contiguous K values starting at
// k + ((l>=16) ? 16 : 0).  One 32-byte load per lane.
__device__ __forceinline__ v16h load_b_frag(const _Float16* __restrict__ M,
                                            int ld, int n0, int k, int lane) {
  const _Float16* p = M + (long long)(n0 + (lane & 15)) * ld
                        + k + ((lane >> 4) << 4);
  return *(const v16h*)p;
}

__device__ __forceinline__ v8f wmma_f16(v16h a, v16h b, v8f c) {
  return __builtin_amdgcn_wmma_f32_16x16x32_f16(
      /*neg_a=*/false, a, /*neg_b=*/false, b,
      /*c_mod=*/(short)0, c, /*reuse_a=*/false, /*reuse_b=*/false);
}

// ---------------------------------------------------------------------------
// Stage 0: w_proj f32 -> f16, layout preserved: [256][768*9] = [N][K]
// ---------------------------------------------------------------------------
__global__ __launch_bounds__(256)
void w_convert_kernel(const float* __restrict__ w, _Float16* __restrict__ w16) {
  int i = blockIdx.x * blockDim.x + threadIdx.x;
  if (i < KD * KCONV) w16[i] = (_Float16)w[i];
}

// ---------------------------------------------------------------------------
// Stage 1: conv3x3 (replicate pad) as WMMA GEMM + L2 channel norm.
// Block = 16 pixels x 256 oc, 8 waves, wave w -> oc tiles {32w, 32w+16}.
// Per k-step (32 of 6912): stage im2col A-tile 16x32 to LDS, 2 WMMAs/wave.
// ---------------------------------------------------------------------------
__global__ __launch_bounds__(256)
void conv_wmma_kernel(const float* __restrict__ cf,
                      const _Float16* __restrict__ w16,
                      const float* __restrict__ bproj,
                      _Float16* __restrict__ Q) {
  __shared__ __align__(32) _Float16 As[16 * LDA];
  __shared__ float ssq[16];

  const int lane = threadIdx.x & 31;
  const int wave = threadIdx.x >> 5;
  const int b    = blockIdx.x / (PN / 16);
  const int n0   = (blockIdx.x % (PN / 16)) * 16;
  const int oc0  = wave * 32;

  // per-thread im2col role: row r = tid>>4, two consecutive kk at (tid&15)*2
  const int r   = threadIdx.x >> 4;
  const int kk0 = (threadIdx.x & 15) * 2;
  const int n   = n0 + r;
  const bool rowvalid = (n < HW);
  const int y = n / HH, x = n % HH;
  int yy[3], xx[3];
  yy[0] = (y > 0) ? y - 1 : 0;  yy[1] = y;  yy[2] = (y < HH - 1) ? y + 1 : HH - 1;
  xx[0] = (x > 0) ? x - 1 : 0;  xx[1] = x;  xx[2] = (x < HH - 1) ? x + 1 : HH - 1;
  const float* fb = cf + (long long)b * CIN * HW;

  v8f acc0 = {}, acc1 = {};
  for (int k0 = 0; k0 < KCONV; k0 += 32) {
    // stage A tile (im2col gather, replicate-pad clamps)
#pragma unroll
    for (int e = 0; e < 2; ++e) {
      int kk = kk0 + e;
      float val = 0.0f;
      if (rowvalid) {
        int kg = k0 + kk;
        int ic = kg / 9;
        int j  = kg - ic * 9;
        int dy = j / 3;
        int dx = j - dy * 3;
        val = fb[(long long)ic * HW + yy[dy] * HH + xx[dx]];
      }
      As[r * LDA + kk] = (_Float16)val;
    }
    __syncthreads();
    v16h a  = load_a_lds(As, LDA, 0, lane);
    v16h b0 = load_b_frag(w16, KCONV, oc0,      k0, lane);
    v16h b1 = load_b_frag(w16, KCONV, oc0 + 16, k0, lane);
    acc0 = wmma_f16(a, b0, acc0);
    acc1 = wmma_f16(a, b1, acc1);
    __syncthreads();
  }

  // bias + block-wide per-row sum of squares (over all 256 oc)
  const float bias0 = bproj[oc0 + (lane & 15)];
  const float bias1 = bproj[oc0 + 16 + (lane & 15)];
#pragma unroll
  for (int v = 0; v < 8; ++v) { acc0[v] += bias0; acc1[v] += bias1; }

  if (threadIdx.x < 16) ssq[threadIdx.x] = 0.0f;
  __syncthreads();
  const int rhi = (lane >> 4) << 3;   // +0 or +8
#pragma unroll
  for (int v = 0; v < 8; ++v)
    atomicAdd(&ssq[rhi + v], acc0[v] * acc0[v] + acc1[v] * acc1[v]);
  __syncthreads();

  const int col0 = oc0 + (lane & 15);
#pragma unroll
  for (int v = 0; v < 8; ++v) {
    int rr = rhi + v;
    int nn = n0 + rr;
    float rinv = rsqrtf(ssq[rr]);
    _Float16* qrow = Q + ((long long)b * PN + nn) * KD;
    bool ok = (nn < HW);
    qrow[col0]      = ok ? (_Float16)(acc0[v] * rinv) : (_Float16)0.0f;
    qrow[col0 + 16] = ok ? (_Float16)(acc1[v] * rinv) : (_Float16)0.0f;
  }
}

// ---------------------------------------------------------------------------
// Stage 2: clipmap f32 -> f16, padded m to PN with zeros.  V[b][c][m]
// ---------------------------------------------------------------------------
__global__ __launch_bounds__(256)
void cm_convert_kernel(const float* __restrict__ cm, _Float16* __restrict__ V) {
  long long i = (long long)blockIdx.x * blockDim.x + threadIdx.x;
  long long total = (long long)NB * CC * PN;
  if (i >= total) return;
  int m = (int)(i % PN);
  long long bc = i / PN;
  V[i] = (m < HW) ? (_Float16)cm[bc * HW + m] : (_Float16)0.0f;
}

// ---------------------------------------------------------------------------
// Stage 3: preds = <f, w_bkg>/||f|| + b_bkg per pixel (1x1 conv, normed feats)
// ---------------------------------------------------------------------------
__global__ __launch_bounds__(256)
void preds_kernel(const float* __restrict__ cf, const float* __restrict__ wbkg,
                  const float* __restrict__ bbkg, float* __restrict__ out) {
  int i = blockIdx.x * blockDim.x + threadIdx.x;
  if (i >= NB * HW) return;
  int b = i / HW, p = i % HW;
  const float* base = cf + (long long)b * CIN * HW + p;
  float dot = 0.0f, ss = 0.0f;
  for (int c = 0; c < CIN; ++c) {
    float f = base[(long long)c * HW];
    dot = fmaf(f, wbkg[c], dot);
    ss  = fmaf(f, f, ss);
  }
  out[i] = dot * rsqrtf(ss) + bbkg[0];
}

// ---------------------------------------------------------------------------
// corrs GEMM: S = Q * Qt (K=256), threshold at 0.2, store P f16 [PN][PN].
// Block = 8 waves; wave w owns a 16x16 tile at m0 = 128*blockIdx.y + 16*w.
// The shared 16x256 A-tile is staged once in LDS (conflict-free stride LDQ).
// Padded Q rows are zero -> padded S entries are 0 -> thresholded to 0.
// ---------------------------------------------------------------------------
__global__ __launch_bounds__(256)
void corrs_kernel(const _Float16* __restrict__ Q, _Float16* __restrict__ P) {
  __shared__ __align__(32) _Float16 Aq[16 * LDQ];
  int lane = threadIdx.x & 31;
  int wave = threadIdx.x >> 5;
  int n0 = blockIdx.x * 16;
  int m0 = blockIdx.y * 128 + wave * 16;

  // cooperative stage of the A-tile: thread t -> row t/16, 16 halves at (t%16)*16
  {
    int r = threadIdx.x >> 4;
    int c = (threadIdx.x & 15) * 16;
    *(v16h*)(Aq + r * LDQ + c) =
        *(const v16h*)(Q + (long long)(n0 + r) * KD + c);
  }
  __syncthreads();

  v8f acc = {};
#pragma unroll
  for (int k = 0; k < KD; k += 32) {
    v16h a = load_a_lds(Aq, LDQ, k, lane);
    v16h b = load_b_frag(Q, KD, m0, k, lane);
    acc = wmma_f16(a, b, acc);
  }
  int col   = m0 + (lane & 15);
  int rbase = n0 + ((lane >> 4) << 3);
#pragma unroll
  for (int v = 0; v < 8; ++v) {
    float s = acc[v];
    P[(long long)(rbase + v) * PN + col] =
        (s < 0.2f) ? (_Float16)0.0f : (_Float16)s;
  }
}

// ---------------------------------------------------------------------------
// rowsum: rs[n] = sum_m P[n][m]  (one wave per row; contiguous lane loads)
// ---------------------------------------------------------------------------
__global__ __launch_bounds__(256)
void rowsum_kernel(const _Float16* __restrict__ P, float* __restrict__ rs) {
  int lane = threadIdx.x & 31;
  int wave = threadIdx.x >> 5;
  int row  = blockIdx.x * 8 + wave;
  if (row >= PN) return;
  const _Float16* pr = P + (long long)row * PN;
  float s = 0.0f;
  for (int m = lane; m < PN; m += 32) s += (float)pr[m];
  __shared__ float red[256];
  red[threadIdx.x] = s;
  __syncthreads();
  if (lane == 0) {
    float t = 0.0f;
#pragma unroll
    for (int j = 0; j < 32; ++j) t += red[wave * 32 + j];
    rs[row] = t;
  }
}

// ---------------------------------------------------------------------------
// refined GEMM: O[n][c] = sum_m P[n][m] * V[c][m] (K=PN), epilogue /(rs+eps),
// scatter to out[c*HW + n].  Block = 8 waves over 128 channels.
// Unrolled x2 with global_prefetch of the next k-chunk (L2-resident operands).
// ---------------------------------------------------------------------------
__global__ __launch_bounds__(256)
void refined_kernel(const _Float16* __restrict__ P, const _Float16* __restrict__ V,
                    const float* __restrict__ rs, float* __restrict__ out) {
  int lane = threadIdx.x & 31;
  int wave = threadIdx.x >> 5;
  int n0 = blockIdx.x * 16;
  int c0 = blockIdx.y * 128 + wave * 16;

  const _Float16* prow = P + (long long)(n0 + (lane & 15)) * PN;
  const _Float16* vrow = V + (long long)(c0 + (lane & 15)) * PN;

  v8f acc = {};
  for (int m = 0; m < PN; m += 64) {   // 38 iterations
    v16h a0 = load_a_frag(P, PN, n0, m,      lane);
    v16h b0 = load_b_frag(V, PN, c0, m,      lane);
    v16h a1 = load_a_frag(P, PN, n0, m + 32, lane);
    v16h b1 = load_b_frag(V, PN, c0, m + 32, lane);
    // prefetch next k-chunk (speculative; OOB at the tail is dropped)
    __builtin_prefetch(prow + m + 64, 0, 3);
    __builtin_prefetch(vrow + m + 64, 0, 3);
    acc = wmma_f16(a0, b0, acc);
    acc = wmma_f16(a1, b1, acc);
  }

  int col   = c0 + (lane & 15);
  int rbase = n0 + ((lane >> 4) << 3);
#pragma unroll
  for (int v = 0; v < 8; ++v) {
    int n = rbase + v;
    if (n < HW) out[(long long)col * HW + n] = acc[v] / (rs[n] + 1e-6f);
  }
}

// ---------------------------------------------------------------------------
extern "C" void kernel_launch(void* const* d_in, const int* in_sizes, int n_in,
                              void* d_out, int out_size, void* d_ws, size_t ws_size,
                              hipStream_t stream) {
  const float* clip_feats = (const float*)d_in[0];
  const float* clipmap    = (const float*)d_in[1];
  const float* w_proj     = (const float*)d_in[2];
  const float* b_proj     = (const float*)d_in[3];
  const float* w_bkg      = (const float*)d_in[4];
  const float* b_bkg      = (const float*)d_in[5];

  float* refined_out = (float*)d_out;                         // [8][512][2401]
  float* preds_out   = refined_out + (long long)NB * CC * HW; // [8][2401]

  // workspace layout (all offsets 32B aligned)
  _Float16* Q   = (_Float16*)d_ws;                   // 8*PN*256    ~10.0 MB
  _Float16* V   = Q + (long long)NB * PN * KD;       // 8*512*PN    ~19.9 MB
  _Float16* P   = V + (long long)NB * CC * PN;       // PN*PN       ~11.8 MB (per-batch reuse)
  _Float16* W16 = P + (long long)PN * PN;            // 256*6912    ~3.5 MB
  float* rowsum = (float*)(W16 + (long long)KD * KCONV);  // PN floats

  w_convert_kernel<<<(KD * KCONV + 255) / 256, 256, 0, stream>>>(w_proj, W16);

  conv_wmma_kernel<<<NB * (PN / 16), 256, 0, stream>>>(clip_feats, W16,
                                                       b_proj, Q);

  long long cmtot = (long long)NB * CC * PN;
  cm_convert_kernel<<<(int)((cmtot + 255) / 256), 256, 0, stream>>>(clipmap, V);

  preds_kernel<<<(NB * HW + 255) / 256, 256, 0, stream>>>(clip_feats, w_bkg,
                                                          b_bkg, preds_out);

  for (int b = 0; b < NB; ++b) {
    corrs_kernel<<<dim3(PN / 16, PN / 128), 256, 0, stream>>>(
        Q + (long long)b * PN * KD, P);
    rowsum_kernel<<<PN / 8, 256, 0, stream>>>(P, rowsum);
    refined_kernel<<<dim3(PN / 16, CC / 128), 256, 0, stream>>>(
        P, V + (long long)b * CC * PN, rowsum,
        refined_out + (long long)b * CC * HW);
  }
}
